// DiffMatch_6468220748547
// MI455X (gfx1250) — matrix operand order repaired and checked by reading, not compiled, not run on hardware
//
#include <hip/hip_runtime.h>
#include <hip/hip_bf16.h>

// ---------------------------------------------------------------------------
// CDNA5 (gfx1250) WMMA types
// ---------------------------------------------------------------------------
typedef __attribute__((ext_vector_type(16))) _Float16 v16h;
typedef __attribute__((ext_vector_type(8)))  float    v8f;

// A fragment: 16x32 f16 tile at (m0, k0) from row-major A with leading dim lda.
// ISA layout (CDNA5 ISA): lanes 0-15 row m0+lane, halves[0..7]=K k0..k0+7,
// halves[8..15]=K k0+16..k0+23 ; lanes 16-31 same rows, K offset +8.
__device__ __forceinline__ v16h load_a_frag(const _Float16* A, int lda,
                                            int m0, int k0, int lane) {
  const int m  = m0 + (lane & 15);
  const int kb = k0 + ((lane & 16) ? 8 : 0);
  const _Float16* p = A + (size_t)m * lda + kb;
  v16h r;
#pragma unroll
  for (int i = 0; i < 8; ++i) r[i] = p[i];
#pragma unroll
  for (int i = 0; i < 8; ++i) r[8 + i] = p[16 + i];
  return r;
}

// B fragment: 32x16 f16 tile at (k0, n0) from TRANSPOSED weights Wt[N][K].
__device__ __forceinline__ v16h load_b_frag(const _Float16* Wt, int ldk,
                                            int n0, int k0, int lane) {
  const int n  = n0 + (lane & 15);
  const int kb = k0 + ((lane & 16) ? 8 : 0);
  const _Float16* p = Wt + (size_t)n * ldk + kb;
  v16h r;
#pragma unroll
  for (int i = 0; i < 8; ++i) r[i] = p[i];
#pragma unroll
  for (int i = 0; i < 8; ++i) r[8 + i] = p[16 + i];
  return r;
}

__device__ __forceinline__ v8f wmma_f16(v16h a, v16h b, v8f c) {
  return __builtin_amdgcn_wmma_f32_16x16x32_f16(false, a, false, b,
                                                (short)0, c, false, false);
}

// ---------------------------------------------------------------------------
// Persistent WMMA GEMM, 128x128 weights staged in LDS, 32-row A tiles staged
// in LDS, B fragments reused across the two 16-row halves (2 accumulators).
// C16 may alias A: A rows are snapshotted into LDS before the epilogue writes.
// ---------------------------------------------------------------------------
__global__ void gemm32(const _Float16* __restrict__ A,   // [M][128] f16
                       const _Float16* __restrict__ Wt,  // [128][128] f16 (N-major)
                       const float* __restrict__ bias,
                       float* __restrict__ C, _Float16* __restrict__ C16,
                       int M, int relu) {
  constexpr int K = 128, N = 128;
  __shared__ __align__(16) _Float16 sW[N * K];   // 32 KB
  __shared__ __align__(16) _Float16 sA[32 * K];  // 8 KB
  const int tid  = threadIdx.x;
  const int wave = tid >> 5, lane = tid & 31;
  // stage weights once per (persistent) block
  {
    const uint4* s = (const uint4*)Wt;
    uint4* d = (uint4*)sW;
    for (int i = tid; i < N * K / 8; i += 256) d[i] = s[i];
  }
  const int tiles = M / 32;
  for (int mt = blockIdx.x; mt < tiles; mt += gridDim.x) {
    __syncthreads();  // previous iteration's sA reads done; sW visible on iter 0
    {
      const uint4* s = (const uint4*)(A + (size_t)mt * 32 * K);
      uint4* d = (uint4*)sA;
      for (int i = tid; i < 32 * K / 8; i += 256) d[i] = s[i];
    }
    if (mt + gridDim.x < tiles)  // speculative prefetch of next A tile
      __builtin_prefetch(A + (size_t)(mt + gridDim.x) * 32 * K, 0, 1);
    __syncthreads();
    v8f acc0 = {}, acc1 = {};
    for (int k0 = 0; k0 < K; k0 += 32) {
      v16h b  = load_b_frag(sW, K, wave * 16, k0, lane);
      v16h a0 = load_a_frag(sA, K, 0,  k0, lane);
      v16h a1 = load_a_frag(sA, K, 16, k0, lane);
      acc0 = wmma_f16(a0, b, acc0);
      acc1 = wmma_f16(a1, b, acc1);
    }
    const int cn = wave * 16 + (lane & 15);
    const int mb = mt * 32 + ((lane & 16) ? 8 : 0);
    const float bv = bias[cn];
#pragma unroll
    for (int r = 0; r < 8; ++r) {
      float v0 = acc0[r] + bv, v1 = acc1[r] + bv;
      if (relu) { v0 = fmaxf(v0, 0.f); v1 = fmaxf(v1, 0.f); }
      const size_t o0 = (size_t)(mb + r) * N + cn;
      const size_t o1 = (size_t)(mb + 16 + r) * N + cn;
      C[o0] = v0; C[o1] = v1;
      if (C16) { C16[o0] = (_Float16)v0; C16[o1] = (_Float16)v1; }
    }
  }
}

// ---------------------------------------------------------------------------
// Fused AGNN edge pass (persistent): e_new = e@A + Ab + hB[src] + hC[dst] +
// tG[graph]; gate=sigmoid(e_new); magg[dst] += gate*hV[src];
// e += relu(e_new); refresh f16 copy. ag_A^T staged in LDS once per block.
// ---------------------------------------------------------------------------
__global__ void edge_fused(const _Float16* __restrict__ e16,
                           float* __restrict__ e,
                           _Float16* __restrict__ e16out,
                           const _Float16* __restrict__ AT,   // [128][128] f16
                           const float* __restrict__ Ab,
                           const float* __restrict__ hB,
                           const float* __restrict__ hC,
                           const float* __restrict__ hV,
                           const float* __restrict__ tG,      // [B][128] this layer
                           float* __restrict__ magg,
                           const int* __restrict__ em0,
                           const int* __restrict__ em1,
                           int E, int nodesPerGraph) {
  constexpr int K = 128, N = 128;
  __shared__ __align__(16) _Float16 sW[N * K];   // 32 KB
  __shared__ __align__(16) _Float16 sA[32 * K];  // 8 KB
  const int tid  = threadIdx.x;
  const int wave = tid >> 5, lane = tid & 31;
  {
    const uint4* s = (const uint4*)AT;
    uint4* d = (uint4*)sW;
    for (int i = tid; i < N * K / 8; i += 256) d[i] = s[i];
  }
  const int tiles = (2 * E) / 32;
  for (int mt = blockIdx.x; mt < tiles; mt += gridDim.x) {
    __syncthreads();
    {
      const uint4* s = (const uint4*)(e16 + (size_t)mt * 32 * K);
      uint4* d = (uint4*)sA;
      for (int i = tid; i < 32 * K / 8; i += 256) d[i] = s[i];
    }
    if (mt + gridDim.x < tiles)
      __builtin_prefetch(e16 + (size_t)(mt + gridDim.x) * 32 * K, 0, 1);
    __syncthreads();
    v8f acc0 = {}, acc1 = {};
    for (int k0 = 0; k0 < K; k0 += 32) {
      v16h b  = load_b_frag(sW, K, wave * 16, k0, lane);
      v16h a0 = load_a_frag(sA, K, 0,  k0, lane);
      v16h a1 = load_a_frag(sA, K, 16, k0, lane);
      acc0 = wmma_f16(a0, b, acc0);
      acc1 = wmma_f16(a1, b, acc1);
    }
    const int cn = wave * 16 + (lane & 15);
    const int mb = mt * 32 + ((lane & 16) ? 8 : 0);
    const float ab = Ab[cn];
#pragma unroll
    for (int half = 0; half < 2; ++half) {
      const v8f& acc = half ? acc1 : acc0;
#pragma unroll
      for (int r = 0; r < 8; ++r) {
        const int j = mb + half * 16 + r;
        int src, dst;
        if (j < E) { src = em0[j];     dst = em1[j];     }
        else       { src = em1[j - E]; dst = em0[j - E]; }
        float en = acc[r] + ab + hB[(size_t)src * 128 + cn]
                 + hC[(size_t)dst * 128 + cn]
                 + tG[(size_t)(src / nodesPerGraph) * 128 + cn];
        const float gate = 1.f / (1.f + __expf(-en));
        atomicAdd(&magg[(size_t)dst * 128 + cn],
                  gate * hV[(size_t)src * 128 + cn]);
        const size_t o = (size_t)j * 128 + cn;
        const float ev = e[o] + fmaxf(en, 0.f);
        e[o] = ev;
        e16out[o] = (_Float16)ev;
      }
    }
  }
}

// ---------------------------------------------------------------------------
// Fused readout per 32-edge tile: m = relu(e@mw1+b1); m = relu(m@mw2+b2);
// ms = m@mw3+b3. A tile + intermediates staged in LDS (24 KB); weight tiles
// come from global (shared by all blocks -> WGP$/L2 resident).
// ---------------------------------------------------------------------------
__global__ void readout_fused(const _Float16* __restrict__ e16,
                              const _Float16* __restrict__ W1T,  // [256][128] f16
                              const float* __restrict__ b1,
                              const _Float16* __restrict__ W2T,  // [128][256] f16
                              const float* __restrict__ b2,
                              const float* __restrict__ w3,
                              const float* __restrict__ b3,
                              float* __restrict__ ms) {
  __shared__ __align__(16) unsigned char smem[32 * 256 * 2];  // 16 KB, reused
  __shared__ __align__(16) _Float16 sA[32 * 128];             // 8 KB
  _Float16* m1 = (_Float16*)smem;  // [32][256] f16
  float*    m2 = (float*)smem;     // [32][128] f32
  const int tid  = threadIdx.x;
  const int wave = tid >> 5, lane = tid & 31;
  const int m0 = blockIdx.x * 32;
  {
    const uint4* s = (const uint4*)(e16 + (size_t)m0 * 128);
    uint4* d = (uint4*)sA;
    for (int i = tid; i < 32 * 128 / 8; i += 256) d[i] = s[i];
  }
  __syncthreads();
  // ---- phase 1: [32x128] @ [128x256] -> LDS f16 ----
  for (int half = 0; half < 2; ++half) {
    const int nt = wave + half * 8;
    v8f acc0 = {}, acc1 = {};
    for (int k0 = 0; k0 < 128; k0 += 32) {
      v16h b  = load_b_frag(W1T, 128, nt * 16, k0, lane);
      v16h a0 = load_a_frag(sA, 128, 0,  k0, lane);
      v16h a1 = load_a_frag(sA, 128, 16, k0, lane);
      acc0 = wmma_f16(a0, b, acc0);
      acc1 = wmma_f16(a1, b, acc1);
    }
    const int cn = nt * 16 + (lane & 15);
    const int rb = (lane & 16) ? 8 : 0;
#pragma unroll
    for (int r = 0; r < 8; ++r) {
      m1[(rb + r) * 256 + cn]      = (_Float16)fmaxf(acc0[r] + b1[cn], 0.f);
      m1[(rb + 16 + r) * 256 + cn] = (_Float16)fmaxf(acc1[r] + b1[cn], 0.f);
    }
  }
  __syncthreads();
  // ---- phase 2: [32x256] @ [256x128], A from LDS ----
  v8f acc0 = {}, acc1 = {};
  for (int k0 = 0; k0 < 256; k0 += 32) {
    v16h b  = load_b_frag(W2T, 256, wave * 16, k0, lane);
    v16h a0 = load_a_frag(m1, 256, 0,  k0, lane);
    v16h a1 = load_a_frag(m1, 256, 16, k0, lane);
    acc0 = wmma_f16(a0, b, acc0);
    acc1 = wmma_f16(a1, b, acc1);
  }
  __syncthreads();  // all m1 reads done before smem is reused for m2
  const int cn = wave * 16 + (lane & 15);
  const int rb = (lane & 16) ? 8 : 0;
#pragma unroll
  for (int r = 0; r < 8; ++r) {
    m2[(rb + r) * 128 + cn]      = fmaxf(acc0[r] + b2[cn], 0.f);
    m2[(rb + 16 + r) * 128 + cn] = fmaxf(acc1[r] + b2[cn], 0.f);
  }
  __syncthreads();
  // ---- phase 3: per-row dot with mw3 ----
  if (tid < 32) {
    float s = 0.f;
    for (int c = 0; c < 128; ++c) s += m2[tid * 128 + c] * w3[c];
    ms[m0 + tid] = s + b3[0];
  }
}

// ---------------------------------------------------------------------------
// Small helper kernels
// ---------------------------------------------------------------------------
__global__ void wt_kernel(const float* __restrict__ w, _Float16* __restrict__ wt,
                          int K, int N) {  // w[K][N] f32 -> wt[N][K] f16
  const int idx = blockIdx.x * blockDim.x + threadIdx.x;
  if (idx >= K * N) return;
  const int n = idx % N, k = idx / N;
  wt[(size_t)n * K + k] = (_Float16)w[(size_t)k * N + n];
}

__global__ void time_kernel(const float* __restrict__ t,
                            const float* __restrict__ tw1, const float* __restrict__ tb1,
                            const float* __restrict__ tw2, const float* __restrict__ tb2,
                            float* __restrict__ temb, int B) {
  const int b = threadIdx.x;
  if (b >= B) return;
  float row[128], u[64];
  const float tv = t[b];
  const float lg = logf(10000.f);
  for (int i = 0; i < 64; ++i) {
    const float a = tv * __expf(-lg * (float)i / 64.f);
    row[i] = __cosf(a);
    row[64 + i] = __sinf(a);
  }
  for (int j = 0; j < 64; ++j) {
    float s = tb1[j];
    for (int i = 0; i < 128; ++i) s += row[i] * tw1[i * 64 + j];
    u[j] = fmaxf(s, 0.f);
  }
  for (int j = 0; j < 64; ++j) {
    float s = tb2[j];
    for (int i = 0; i < 64; ++i) s += u[i] * tw2[i * 64 + j];
    temb[b * 64 + j] = s;
  }
}

__global__ void tg_kernel(const float* __restrict__ temb,
                          const float* __restrict__ agT, const float* __restrict__ agTb,
                          float* __restrict__ tg, int B) {
  const int idx = blockIdx.x * blockDim.x + threadIdx.x;
  if (idx >= 3 * B * 128) return;
  const int j = idx & 127;
  const int b = (idx >> 7) % B;
  const int l = idx / (B * 128);
  float s = agTb[l * 128 + j];
  for (int i = 0; i < 64; ++i)
    s += temb[b * 64 + i] * agT[((size_t)l * 64 + i) * 128 + j];
  tg[idx] = s;
}

__global__ void pe_kernel(const float* __restrict__ attr, _Float16* __restrict__ pe,
                          int E) {
  const int idx = blockIdx.x * blockDim.x + threadIdx.x;
  if (idx >= 2 * E * 128) return;
  const int c = idx & 127;
  const int j = idx >> 7;
  const float a = attr[j < E ? j : j - E];
  const float d = __expf(logf(10000.f) * (float)(c >> 1) / 64.f);
  const float pos = a / d;
  pe[idx] = (_Float16)((c & 1) ? __cosf(pos) : __sinf(pos));
}

__global__ void copy_h(const float* __restrict__ src, float* __restrict__ h,
                       _Float16* __restrict__ h16, int n) {
  const int i = blockIdx.x * blockDim.x + threadIdx.x;
  if (i >= n) return;
  const float v = src[i];
  h[i] = v; h16[i] = (_Float16)v;
}

__global__ void gin_pre(const float* __restrict__ h, float* __restrict__ z,
                        const float* __restrict__ eps, int l, int n) {
  const int i = blockIdx.x * blockDim.x + threadIdx.x;
  if (i >= n) return;
  z[i] = (1.f + eps[l]) * h[i];
}

__global__ void gin_scatter(const float* __restrict__ h,
                            const int* __restrict__ gsrc, const int* __restrict__ gdst,
                            float* __restrict__ z, int EG) {
  const int idx = blockIdx.x * blockDim.x + threadIdx.x;
  if (idx >= EG * 128) return;
  const int edge = idx >> 7, c = idx & 127;
  atomicAdd(&z[(size_t)gdst[edge] * 128 + c], h[(size_t)gsrc[edge] * 128 + c]);
}

__global__ void f32tof16(const float* __restrict__ a, _Float16* __restrict__ b, int n) {
  const int i = blockIdx.x * blockDim.x + threadIdx.x;
  if (i < n) b[i] = (_Float16)a[i];
}

__global__ void zero_kernel(float* __restrict__ p, int n) {
  const int i = blockIdx.x * blockDim.x + threadIdx.x;
  if (i < n) p[i] = 0.f;
}

// GraphNorm over fixed 48-node segments (bn_batch == node/48), then ReLU.
__global__ void gnorm_kernel(const float* __restrict__ z,
                             const float* __restrict__ gw, const float* __restrict__ gb,
                             const float* __restrict__ gms,
                             float* __restrict__ h, _Float16* __restrict__ h16) {
  const int seg = blockIdx.x;
  const int c   = threadIdx.x;
  const int base = seg * 48;
  float s = 0.f;
  for (int n = 0; n < 48; ++n) s += z[(size_t)(base + n) * 128 + c];
  const float mean = s * (1.f / 48.f) * gms[c];
  float v = 0.f;
  for (int n = 0; n < 48; ++n) {
    const float zc = z[(size_t)(base + n) * 128 + c] - mean;
    v += zc * zc;
  }
  const float inv = rsqrtf(v * (1.f / 48.f) + 1e-5f);
  const float w = gw[c], b = gb[c];
  for (int n = 0; n < 48; ++n) {
    const float zc = z[(size_t)(base + n) * 128 + c] - mean;
    const float o = fmaxf(w * zc * inv + b, 0.f);
    h[(size_t)(base + n) * 128 + c] = o;
    h16[(size_t)(base + n) * 128 + c] = (_Float16)o;
  }
}

__global__ void hupd_kernel(float* __restrict__ h, _Float16* __restrict__ h16,
                            const float* __restrict__ hU, const float* __restrict__ magg,
                            int n) {
  const int i = blockIdx.x * blockDim.x + threadIdx.x;
  if (i >= n) return;
  const float v = h[i] + fmaxf(hU[i] + magg[i], 0.f);
  h[i] = v; h16[i] = (_Float16)v;
}

__global__ void combine_kernel(const float* __restrict__ ms, float* __restrict__ out,
                               int E) {
  const int i = blockIdx.x * blockDim.x + threadIdx.x;
  if (i < E) out[i] = ms[i] + ms[i + E];
}

// ---------------------------------------------------------------------------
// Host-side orchestration
// ---------------------------------------------------------------------------
extern "C" void kernel_launch(void* const* d_in, const int* in_sizes, int n_in,
                              void* d_out, int out_size, void* d_ws, size_t ws_size,
                              hipStream_t stream) {
  (void)n_in; (void)out_size; (void)ws_size;
  const float* graph_x = (const float*)d_in[0];
  const float* t_in    = (const float*)d_in[1];
  const float* noise   = (const float*)d_in[2];
  const int*   gei     = (const int*)d_in[3];
  const int*   emap    = (const int*)d_in[4];
  const float* gin_eps = (const float*)d_in[7];
  const float* gin_w1  = (const float*)d_in[8];
  const float* gin_b1  = (const float*)d_in[9];
  const float* gin_w2  = (const float*)d_in[10];
  const float* gin_b2  = (const float*)d_in[11];
  const float* gn_w    = (const float*)d_in[12];
  const float* gn_b    = (const float*)d_in[13];
  const float* gn_ms   = (const float*)d_in[14];
  const float* ag_U  = (const float*)d_in[15]; const float* ag_Ub = (const float*)d_in[16];
  const float* ag_V  = (const float*)d_in[17]; const float* ag_Vb = (const float*)d_in[18];
  const float* ag_A  = (const float*)d_in[19]; const float* ag_Ab = (const float*)d_in[20];
  const float* ag_Bm = (const float*)d_in[21]; const float* ag_Bb = (const float*)d_in[22];
  const float* ag_C  = (const float*)d_in[23]; const float* ag_Cb = (const float*)d_in[24];
  const float* ag_T  = (const float*)d_in[25]; const float* ag_Tb = (const float*)d_in[26];
  const float* tw1 = (const float*)d_in[27]; const float* tb1 = (const float*)d_in[28];
  const float* tw2 = (const float*)d_in[29]; const float* tb2 = (const float*)d_in[30];
  const float* ew  = (const float*)d_in[31]; const float* eb  = (const float*)d_in[32];
  const float* mw1 = (const float*)d_in[33]; const float* mb1 = (const float*)d_in[34];
  const float* mw2 = (const float*)d_in[35]; const float* mb2 = (const float*)d_in[36];
  const float* mw3 = (const float*)d_in[37]; const float* mb3 = (const float*)d_in[38];

  const int E   = in_sizes[2];        // 147456
  const int E2  = 2 * E;              // 294912
  const int Nn  = in_sizes[0] / 128;  // 6144 nodes
  const int EG  = in_sizes[3] / 2;    // 98304 intra-graph edges
  const int Bn  = in_sizes[1];        // 64 graph pairs
  const int NPG2 = Nn / Bn;           // 96 nodes per graph-pair

  // ---- workspace partition (256B aligned) ----
  char* p = (char*)d_ws;
  auto take = [&](size_t bytes) {
    char* q = p; p += (bytes + 255) & ~(size_t)255; return q;
  };
  _Float16* ewT  = (_Float16*)take((size_t)128 * 128 * 2);
  _Float16* gw1T = (_Float16*)take((size_t)3 * 128 * 128 * 2);
  _Float16* gw2T = (_Float16*)take((size_t)3 * 128 * 128 * 2);
  _Float16* uT   = (_Float16*)take((size_t)3 * 128 * 128 * 2);
  _Float16* vT   = (_Float16*)take((size_t)3 * 128 * 128 * 2);
  _Float16* aT   = (_Float16*)take((size_t)3 * 128 * 128 * 2);
  _Float16* bT   = (_Float16*)take((size_t)3 * 128 * 128 * 2);
  _Float16* cT   = (_Float16*)take((size_t)3 * 128 * 128 * 2);
  _Float16* m1T  = (_Float16*)take((size_t)256 * 128 * 2);
  _Float16* m2T  = (_Float16*)take((size_t)128 * 256 * 2);
  float* temb = (float*)take((size_t)Bn * 64 * 4);
  float* tg   = (float*)take((size_t)3 * Bn * 128 * 4);
  float* h    = (float*)take((size_t)Nn * 128 * 4);
  _Float16* h16 = (_Float16*)take((size_t)Nn * 128 * 2);
  float* nf1  = (float*)take((size_t)Nn * 128 * 4);
  float* nf2  = (float*)take((size_t)Nn * 128 * 4);
  _Float16* nh1 = (_Float16*)take((size_t)Nn * 128 * 2);
  _Float16* nh2 = (_Float16*)take((size_t)Nn * 128 * 2);
  float* hBv  = (float*)take((size_t)Nn * 128 * 4);
  float* hCv  = (float*)take((size_t)Nn * 128 * 4);
  float* hVv  = (float*)take((size_t)Nn * 128 * 4);
  float* hUv  = (float*)take((size_t)Nn * 128 * 4);
  float* magg = (float*)take((size_t)Nn * 128 * 4);
  float* e    = (float*)take((size_t)E2 * 128 * 4);      // 151 MB (L2 resident)
  _Float16* e16 = (_Float16*)take((size_t)E2 * 128 * 2); // 75 MB
  float* ms   = (float*)take((size_t)E2 * 4);

  auto cdiv = [](int a, int b) { return (a + b - 1) / b; };
  const int T = 256;
  const int nodeGrid = Nn / 32;                       // 192 persistent blocks
  const int edgeGrid = (E2 / 32 < 2048) ? E2 / 32 : 2048;

  // ---- weight conversion (fp32 -> f16, transposed) ----
  wt_kernel<<<cdiv(128 * 128, T), T, 0, stream>>>(ew, ewT, 128, 128);
  for (int l = 0; l < 3; ++l) {
    const size_t o = (size_t)l * 128 * 128;
    wt_kernel<<<cdiv(128 * 128, T), T, 0, stream>>>(gin_w1 + o, gw1T + o, 128, 128);
    wt_kernel<<<cdiv(128 * 128, T), T, 0, stream>>>(gin_w2 + o, gw2T + o, 128, 128);
    wt_kernel<<<cdiv(128 * 128, T), T, 0, stream>>>(ag_U  + o, uT + o, 128, 128);
    wt_kernel<<<cdiv(128 * 128, T), T, 0, stream>>>(ag_V  + o, vT + o, 128, 128);
    wt_kernel<<<cdiv(128 * 128, T), T, 0, stream>>>(ag_A  + o, aT + o, 128, 128);
    wt_kernel<<<cdiv(128 * 128, T), T, 0, stream>>>(ag_Bm + o, bT + o, 128, 128);
    wt_kernel<<<cdiv(128 * 128, T), T, 0, stream>>>(ag_C  + o, cT + o, 128, 128);
  }
  wt_kernel<<<cdiv(128 * 256, T), T, 0, stream>>>(mw1, m1T, 128, 256);
  wt_kernel<<<cdiv(256 * 128, T), T, 0, stream>>>(mw2, m2T, 256, 128);

  // ---- time embedding + per-layer time projections ----
  time_kernel<<<1, 64, 0, stream>>>(t_in, tw1, tb1, tw2, tb2, temb, Bn);
  tg_kernel<<<cdiv(3 * Bn * 128, T), T, 0, stream>>>(temb, ag_T, ag_Tb, tg, Bn);

  // ---- edge positional embedding -> e = pe @ ew + eb (e16 rewritten in place) ----
  pe_kernel<<<cdiv(E2 * 128, T), T, 0, stream>>>(noise, e16, E);
  copy_h<<<cdiv(Nn * 128, T), T, 0, stream>>>(graph_x, h, h16, Nn * 128);
  gemm32<<<edgeGrid, T, 0, stream>>>(e16, ewT, eb, e, e16, E2, 0);

  // ---- layers ----
  for (int l = 0; l < 3; ++l) {
    const size_t wo = (size_t)l * 128 * 128;
    const size_t bo = (size_t)l * 128;
    // GIN
    gin_pre<<<cdiv(Nn * 128, T), T, 0, stream>>>(h, nf1, gin_eps, l, Nn * 128);
    gin_scatter<<<cdiv(EG * 128, T), T, 0, stream>>>(h, gei, gei + EG, nf1, EG);
    f32tof16<<<cdiv(Nn * 128, T), T, 0, stream>>>(nf1, nh1, Nn * 128);
    gemm32<<<nodeGrid, T, 0, stream>>>(nh1, gw1T + wo, gin_b1 + bo, nf2, nh2, Nn, 1);
    gemm32<<<nodeGrid, T, 0, stream>>>(nh2, gw2T + wo, gin_b2 + bo, nf1, nullptr, Nn, 0);
    // GraphNorm + ReLU
    gnorm_kernel<<<Nn / 48, 128, 0, stream>>>(nf1, gn_w + bo, gn_b + bo, gn_ms + bo, h, h16);
    // node projections for AGNN
    gemm32<<<nodeGrid, T, 0, stream>>>(h16, bT + wo, ag_Bb + bo, hBv, nullptr, Nn, 0);
    gemm32<<<nodeGrid, T, 0, stream>>>(h16, cT + wo, ag_Cb + bo, hCv, nullptr, Nn, 0);
    gemm32<<<nodeGrid, T, 0, stream>>>(h16, vT + wo, ag_Vb + bo, hVv, nullptr, Nn, 0);
    gemm32<<<nodeGrid, T, 0, stream>>>(h16, uT + wo, ag_Ub + bo, hUv, nullptr, Nn, 0);
    // gated edge pass (fused GEMM + gate + scatter + residual)
    zero_kernel<<<cdiv(Nn * 128, T), T, 0, stream>>>(magg, Nn * 128);
    edge_fused<<<edgeGrid, T, 0, stream>>>(e16, e, e16, aT + wo, ag_Ab + bo,
                                           hBv, hCv, hVv, tg + (size_t)l * Bn * 128,
                                           magg, emap, emap + E, E, NPG2);
    // h = h + relu(hU + magg)
    hupd_kernel<<<cdiv(Nn * 128, T), T, 0, stream>>>(h, h16, hUv, magg, Nn * 128);
  }

  // ---- readout + symmetrize ----
  readout_fused<<<E2 / 32, T, 0, stream>>>(e16, m1T, mb1, m2T, mb2, mw3, mb3, ms);
  combine_kernel<<<cdiv(E, T), T, 0, stream>>>(ms, (float*)d_out, E);
}
